// Net_27504970563736
// MI455X (gfx1250) — matrix operand order repaired and checked by reading, not compiled, not run on hardware
//
#include <hip/hip_runtime.h>
#include <hip/hip_bf16.h>
#include <math.h>

// ---------------- dims ----------------
#define B_    8
#define L_    1024
#define IN_   16
#define H_    256
#define NLAYER 2
#define ED    512          // D_INNER
#define NST   16           // D_STATE
#define DTR   16           // DT_RANK
#define DCONV 4
#define ROWS  (B_*L_)      // 8192
#define HEADK (H_*L_)      // 262144

typedef __attribute__((ext_vector_type(16))) __bf16 v16bf;
typedef __attribute__((ext_vector_type(8)))  float  v8f;

__device__ __forceinline__ float silu_f(float x)     { return x / (1.f + __expf(-x)); }
__device__ __forceinline__ float softplus_f(float x) { return (x > 20.f) ? x : logf(1.f + __expf(x)); }

// =====================================================================
// Generic WMMA GEMM:  C[M,N] = act(A[M,K] * W[N,K]^T + bias[N]) + resid[M,N]
// A fp32 (row stride lda), W fp32 row-major [N,K], C fp32 (row stride ldc).
// bf16 compute (v_wmma_f32_16x16x32_bf16), fp32 accumulate.
// 256 threads = 8 waves (4x2), tile 128x128, K-step 32, double-buffered LDS.
// =====================================================================
#define TM 128
#define TN 128
#define TK 32
#define LDA_S 36   // padded LDS stride (bf16) for A tile: 72B/row -> 8B-aligned chunks
#define LDB_S 132  // padded LDS stride (bf16) for transposed B tile: 264B/row

// ---- global -> registers (4x float4 each for A and W) ----
template<bool CHECK>
__device__ __forceinline__ void stage_gload(
    const float* __restrict__ A, int lda, int M, int K,
    const float* __restrict__ W, int N,
    int m0, int n0, int k0, int tid,
    float4 (&areg)[4], float4 (&breg)[4])
{
    #pragma unroll
    for (int t = 0; t < 4; ++t) {
        const int i = (tid + t * 256) << 2;   // element index in 128x32 tile
        const int r = i >> 5;                 // tile row (M or N)
        const int c = i & 31;                 // tile col (K), multiple of 4
        const int gk = k0 + c;
        const int gm = m0 + r;
        if (!CHECK || (gm < M && gk < K))
            areg[t] = *(const float4*)&A[(size_t)gm * lda + gk];
        else
            areg[t] = make_float4(0.f, 0.f, 0.f, 0.f);
        const int gn = n0 + r;
        if (!CHECK || (gn < N && gk < K))
            breg[t] = *(const float4*)&W[(size_t)gn * K + gk];
        else
            breg[t] = make_float4(0.f, 0.f, 0.f, 0.f);
    }
}

// ---- registers -> LDS (A packed b64 stores; B transposed b16 stores) ----
__device__ __forceinline__ void stage_store(
    __bf16* __restrict__ As, __bf16* __restrict__ Bs, int tid,
    const float4 (&areg)[4], const float4 (&breg)[4])
{
    #pragma unroll
    for (int t = 0; t < 4; ++t) {
        const int i = (tid + t * 256) << 2;
        const int r = i >> 5;
        const int c = i & 31;
        union { __bf16 h[4]; unsigned long long u; } pk;
        pk.h[0] = (__bf16)areg[t].x; pk.h[1] = (__bf16)areg[t].y;
        pk.h[2] = (__bf16)areg[t].z; pk.h[3] = (__bf16)areg[t].w;
        *(unsigned long long*)&As[r * LDA_S + c] = pk.u;      // 8B-aligned
        Bs[(c + 0) * LDB_S + r] = (__bf16)breg[t].x;          // transpose: Bs[k][n]
        Bs[(c + 1) * LDB_S + r] = (__bf16)breg[t].y;
        Bs[(c + 2) * LDB_S + r] = (__bf16)breg[t].z;
        Bs[(c + 3) * LDB_S + r] = (__bf16)breg[t].w;
    }
}

// ---- fragment loads + 2x4 WMMAs for one K-step ----
__device__ __forceinline__ void do_wmma(
    const __bf16* __restrict__ As, const __bf16* __restrict__ Bs,
    int lane, int wm, int wn, v8f (&acc)[2][4])
{
    const int arow  = lane & 15;
    const int khalf = (lane >> 4) * 8;   // lanes 0-15: K{0..7,16..23}; 16-31: K{8..15,24..31}
    v16bf afrag[2];
    #pragma unroll
    for (int i = 0; i < 2; ++i) {
        const __bf16* p = &As[(wm * 32 + i * 16 + arow) * LDA_S + khalf];
        #pragma unroll
        for (int e = 0; e < 8; ++e) { afrag[i][e] = p[e]; afrag[i][8 + e] = p[16 + e]; }
    }
    v16bf bfrag[4];                      // lane L holds B column N=L%16, K in same order as A
    #pragma unroll
    for (int j = 0; j < 4; ++j) {
        const __bf16* p = &Bs[khalf * LDB_S + wn * 64 + j * 16 + arow];
        #pragma unroll
        for (int e = 0; e < 8; ++e) { bfrag[j][e] = p[e * LDB_S]; bfrag[j][8 + e] = p[(16 + e) * LDB_S]; }
    }
    #pragma unroll
    for (int i = 0; i < 2; ++i)
        #pragma unroll
        for (int j = 0; j < 4; ++j)
            acc[i][j] = __builtin_amdgcn_wmma_f32_16x16x32_bf16(
                false, afrag[i], false, bfrag[j], (short)0, acc[i][j], false, false);
}

__global__ __launch_bounds__(256)
void gemm_wmma_bf16(const float* __restrict__ A, int lda,
                    const float* __restrict__ W,
                    const float* __restrict__ bias,
                    const float* __restrict__ resid,
                    float* __restrict__ C, int ldc,
                    int M, int N, int K, int act)
{
    __shared__ __bf16 As[2][TM * LDA_S];
    __shared__ __bf16 Bs[2][TK * LDB_S];

    const int tid  = threadIdx.x;
    const int lane = tid & 31;
    const int wave = tid >> 5;
    const int wm   = wave & 3;          // 4 wave-rows of 32
    const int wn   = wave >> 2;         // 2 wave-cols of 64
    const int m0   = blockIdx.x * TM;
    const int n0   = blockIdx.y * TN;

    v8f acc[2][4] = {};

    const int  KT     = (K + TK - 1) / TK;
    const bool edgeMN = (m0 + TM > M) || (n0 + TN > N);

    float4 areg[4], breg[4];
    {
        const bool e = edgeMN || (TK > K);
        if (e) stage_gload<true >(A, lda, M, K, W, N, m0, n0, 0, tid, areg, breg);
        else   stage_gload<false>(A, lda, M, K, W, N, m0, n0, 0, tid, areg, breg);
    }
    stage_store(As[0], Bs[0], tid, areg, breg);
    __syncthreads();

    for (int kt = 0; kt < KT; ++kt) {
        const int  cur  = kt & 1;
        const bool more = (kt + 1 < KT);
        if (more) {   // issue next tile's global loads before compute
            const int  k0 = (kt + 1) * TK;
            const bool e  = edgeMN || (k0 + TK > K);
            if (e) stage_gload<true >(A, lda, M, K, W, N, m0, n0, k0, tid, areg, breg);
            else   stage_gload<false>(A, lda, M, K, W, N, m0, n0, k0, tid, areg, breg);
        }
        do_wmma(As[cur], Bs[cur], lane, wm, wn, acc);
        if (more) {
            stage_store(As[1 - cur], Bs[1 - cur], tid, areg, breg);
            __syncthreads();
        }
    }

    // ---- epilogue: lane<16 -> rows 0..7, lane>=16 -> rows 8..15 of each 16x16
    const int ncol = lane & 15;
    const int mrow = (lane >> 4) * 8;
    #pragma unroll
    for (int i = 0; i < 2; ++i) {
        #pragma unroll
        for (int j = 0; j < 4; ++j) {
            const int cm0 = m0 + wm * 32 + i * 16 + mrow;
            const int cn  = n0 + wn * 64 + j * 16 + ncol;
            if (cn < N) {
                const float bv = bias ? bias[cn] : 0.f;
                #pragma unroll
                for (int r = 0; r < 8; ++r) {
                    const int cm = cm0 + r;
                    if (cm < M) {
                        float v = acc[i][j][r] + bv;
                        if (act == 1)      v = fmaxf(v, 0.f);
                        else if (act == 2) v = softplus_f(v);
                        if (resid) v += resid[(size_t)cm * ldc + cn];
                        C[(size_t)cm * ldc + cn] = v;
                    }
                }
            }
        }
    }
}

// =====================================================================
// RMSNorm over H=256; one block per row
// =====================================================================
__global__ __launch_bounds__(256)
void rmsnorm_kernel(const float* __restrict__ h, const float* __restrict__ w,
                    float* __restrict__ out)
{
    __shared__ float red[256];
    int row = blockIdx.x, t = threadIdx.x;
    float v = h[(size_t)row * H_ + t];
    red[t] = v * v;
    __syncthreads();
    for (int s = 128; s > 0; s >>= 1) {
        if (t < s) red[t] += red[t + s];
        __syncthreads();
    }
    float scale = rsqrtf(red[0] * (1.f / H_) + 1e-5f);
    out[(size_t)row * H_ + t] = v * scale * w[t];
}

// =====================================================================
// Depthwise causal conv1d (k=4) over xi = xz[:,:,0:512], + bias, + silu
// =====================================================================
__global__ __launch_bounds__(256)
void conv_silu_kernel(const float* __restrict__ xz, const float* __restrict__ cw,
                      const float* __restrict__ cb, float* __restrict__ xic)
{
    size_t idx = (size_t)blockIdx.x * 256 + threadIdx.x;
    if (idx >= (size_t)ROWS * ED) return;
    int d = idx & (ED - 1);
    int l = (idx >> 9) & (L_ - 1);
    int b = (int)(idx >> 19);
    float acc = cb[d];
    #pragma unroll
    for (int j = 0; j < DCONV; ++j) {
        int li = l - (DCONV - 1) + j;
        if (li >= 0)
            acc += xz[((size_t)(b * L_ + li)) * (2 * ED) + d] * cw[d * DCONV + j];
    }
    xic[idx] = silu_f(acc);
}

// =====================================================================
// Selective scan, one lane per (b,d), 16 states in registers.
// Fuses y = (scan + xi*Dp) * silu(z).  grid = (ED/256, B_)
// =====================================================================
__global__ __launch_bounds__(256)
void scan_kernel(const float* __restrict__ delta, const float* __restrict__ xic,
                 const float* __restrict__ dbc,   const float* __restrict__ xz,
                 const float* __restrict__ A_log, const float* __restrict__ Dp,
                 float* __restrict__ yg)
{
    int d = blockIdx.x * 256 + threadIdx.x;
    int b = blockIdx.y;
    float a[NST], st[NST];
    #pragma unroll
    for (int n = 0; n < NST; ++n) { a[n] = -__expf(A_log[d * NST + n]); st[n] = 0.f; }
    float Dpd = Dp[d];
    for (int l = 0; l < L_; ++l) {
        size_t row = (size_t)(b * L_ + l);
        float dl = delta[row * ED + d];
        float xv = xic[row * ED + d];
        float zv = xz[row * (2 * ED) + ED + d];
        const float* bc = &dbc[row * (DTR + 2 * NST)];
        float dx = dl * xv;
        float y  = 0.f;
        #pragma unroll
        for (int n = 0; n < NST; ++n) {
            float dA = __expf(dl * a[n]);
            st[n] = dA * st[n] + dx * bc[DTR + n];        // B
            y    += st[n] * bc[DTR + NST + n];            // C
        }
        y += xv * Dpd;
        yg[row * ED + d] = y * silu_f(zv);
    }
}

// =====================================================================
// head1: [8, 262144] x w1[256, 262144]^T + b1, relu.  M=8 -> HBM-bound
// reduction, one block per output column n.
// =====================================================================
__global__ __launch_bounds__(256)
void head1_kernel(const float* __restrict__ hflat, const float* __restrict__ w1,
                  const float* __restrict__ b1, float* __restrict__ hh)
{
    __shared__ float red[B_][256];
    int n = blockIdx.x, t = threadIdx.x;
    const float* wrow = w1 + (size_t)n * HEADK;
    float acc[B_] = {};
    for (int k = t; k < HEADK; k += 256) {
        if (k + 4096 < HEADK) __builtin_prefetch(&wrow[k + 4096], 0, 1);
        float wv = wrow[k];
        #pragma unroll
        for (int b = 0; b < B_; ++b) acc[b] += wv * hflat[(size_t)b * HEADK + k];
    }
    #pragma unroll
    for (int b = 0; b < B_; ++b) red[b][t] = acc[b];
    __syncthreads();
    for (int s = 128; s > 0; s >>= 1) {
        if (t < s) {
            #pragma unroll
            for (int b = 0; b < B_; ++b) red[b][t] += red[b][t + s];
        }
        __syncthreads();
    }
    if (t < B_) hh[t * H_ + n] = fmaxf(red[t][0] + b1[n], 0.f);
}

// =====================================================================
// head2: [8,256] x w2[1,256]^T + b2 -> out[8]
// =====================================================================
__global__ void head2_kernel(const float* __restrict__ hh, const float* __restrict__ w2,
                             const float* __restrict__ b2, float* __restrict__ out)
{
    int b = threadIdx.x;
    if (b < B_) {
        float acc = b2[0];
        for (int k = 0; k < H_; ++k) acc += hh[b * H_ + k] * w2[k];
        out[b] = acc;
    }
}

// =====================================================================
extern "C" void kernel_launch(void* const* d_in, const int* in_sizes, int n_in,
                              void* d_out, int out_size, void* d_ws, size_t ws_size,
                              hipStream_t stream)
{
    (void)in_sizes; (void)n_in; (void)out_size; (void)ws_size;
    const float* x      = (const float*)d_in[0];
    const float* emb_w  = (const float*)d_in[1];
    const float* emb_b  = (const float*)d_in[2];
    // layers: 10 tensors each starting at index 3:
    // norm_w, in_proj_w, conv_w, conv_b, x_proj_w, dt_proj_w, dt_proj_b, A_log, Dp, out_proj_w
    const int LBASE = 3;
    const float* head_w1 = (const float*)d_in[LBASE + NLAYER * 10 + 0];
    const float* head_b1 = (const float*)d_in[LBASE + NLAYER * 10 + 1];
    const float* head_w2 = (const float*)d_in[LBASE + NLAYER * 10 + 2];
    const float* head_b2 = (const float*)d_in[LBASE + NLAYER * 10 + 3];

    // ---- workspace carve (floats) ----
    float* ws = (float*)d_ws;
    float* h0    = ws;                       ws += (size_t)ROWS * H_;
    float* h1    = ws;                       ws += (size_t)ROWS * H_;
    float* xn    = ws;                       ws += (size_t)ROWS * H_;
    float* xz    = ws;                       ws += (size_t)ROWS * 2 * ED;
    float* xic   = ws;                       ws += (size_t)ROWS * ED;
    float* dbc   = ws;                       ws += (size_t)ROWS * (DTR + 2 * NST);
    float* delta = ws;                       ws += (size_t)ROWS * ED;
    float* yg    = ws;                       ws += (size_t)ROWS * ED;
    float* hh    = ws;                       ws += (size_t)B_ * H_;

    dim3 blk(256);
    #define GG(Mv, Nv) dim3(((Mv) + TM - 1) / TM, ((Nv) + TN - 1) / TN)

    // embedding: h0 = x @ emb_w^T + emb_b     [8192,16]x[256,16]^T
    gemm_wmma_bf16<<<GG(ROWS, H_), blk, 0, stream>>>(
        x, IN_, emb_w, emb_b, nullptr, h0, H_, ROWS, H_, IN_, 0);

    float* hin = h0;
    float* hout = h1;
    for (int ly = 0; ly < NLAYER; ++ly) {
        const float* const* lp = (const float* const*)(d_in + LBASE + ly * 10);
        const float* norm_w    = lp[0];
        const float* in_proj_w = lp[1];
        const float* conv_w    = lp[2];
        const float* conv_b    = lp[3];
        const float* x_proj_w  = lp[4];
        const float* dt_proj_w = lp[5];
        const float* dt_proj_b = lp[6];
        const float* A_log     = lp[7];
        const float* Dp        = lp[8];
        const float* out_proj_w= lp[9];

        rmsnorm_kernel<<<ROWS, blk, 0, stream>>>(hin, norm_w, xn);

        // in_proj: xz = xn @ in_proj_w^T      [8192,256]x[1024,256]^T
        gemm_wmma_bf16<<<GG(ROWS, 2 * ED), blk, 0, stream>>>(
            xn, H_, in_proj_w, nullptr, nullptr, xz, 2 * ED, ROWS, 2 * ED, H_, 0);

        // depthwise causal conv + silu on xi half
        conv_silu_kernel<<<(int)(((size_t)ROWS * ED) / 256), blk, 0, stream>>>(
            xz, conv_w, conv_b, xic);

        // x_proj: dbc = xic @ x_proj_w^T      [8192,512]x[48,512]^T
        gemm_wmma_bf16<<<GG(ROWS, DTR + 2 * NST), blk, 0, stream>>>(
            xic, ED, x_proj_w, nullptr, nullptr, dbc, DTR + 2 * NST,
            ROWS, DTR + 2 * NST, ED, 0);

        // dt_proj: delta = softplus(dbc[:, :16] @ dt_proj_w^T + b)
        gemm_wmma_bf16<<<GG(ROWS, ED), blk, 0, stream>>>(
            dbc, DTR + 2 * NST, dt_proj_w, dt_proj_b, nullptr, delta, ED,
            ROWS, ED, DTR, 2);

        // selective scan + gating
        scan_kernel<<<dim3(ED / 256, B_), blk, 0, stream>>>(
            delta, xic, dbc, xz, A_log, Dp, yg);

        // out_proj + residual: hout = yg @ out_proj_w^T + hin
        gemm_wmma_bf16<<<GG(ROWS, H_), blk, 0, stream>>>(
            yg, ED, out_proj_w, nullptr, hin, hout, H_, ROWS, H_, ED, 0);

        float* t = hin; hin = hout; hout = t;
    }

    head1_kernel<<<H_, blk, 0, stream>>>(hin, head_w1, head_b1, hh);
    head2_kernel<<<1, 32, 0, stream>>>(hh, head_w2, head_b2, (float*)d_out);
}
